// MultiHeadAttention_23304492548097
// MI455X (gfx1250) — matrix operand order repaired and checked by reading, not compiled, not run on hardware
//
#include <hip/hip_runtime.h>
#include <hip/hip_bf16.h>

typedef unsigned short u16;
typedef unsigned int   u32;
typedef unsigned long long u64;
typedef __attribute__((ext_vector_type(4)))  u32    v4u;
typedef __attribute__((ext_vector_type(8)))  u32    v8u;
typedef __attribute__((ext_vector_type(8)))  float  v8f;
typedef __attribute__((ext_vector_type(16))) __bf16 v16bf;

// Problem constants
#define BB 2
#define SS 2048
#define DD 4096
#define HH 32
#define HDD 128
#define HALFD 64
#define LOG2E 1.4426950408889634f

// LDS tile strides (padded for bank-conflict-free fragment reads)
#define KLD 136   // 128 + 8 u16 pad  -> row stride 68 dwords (gcd(68,64)=4, 4*l15 distinct)
#define VLD 40    // 32 + 8 u16 pad   -> row stride 20 dwords (20*l15 mod 64 distinct)

// ---------------- bf16 helpers ----------------
__device__ inline u16 f2bf(float f) {
    u32 u = __builtin_bit_cast(u32, f);
    u32 r = u + 0x7FFFu + ((u >> 16) & 1u);   // round-to-nearest-even
    return (u16)(r >> 16);
}
__device__ inline float bf2f(u16 h) {
    u32 u = ((u32)h) << 16;
    return __builtin_bit_cast(float, u);
}
__device__ inline u32 pack2bf(float lo, float hi) {
    return (u32)f2bf(lo) | ((u32)f2bf(hi) << 16);
}

// ---------------- WMMA fragment loaders ----------------
// A-fragment 16x32 bf16 (row per lane): dwords 0..3 K=8*half+{0..7}, dwords 4..7 K=16+8*half+{0..7}
__device__ inline v16bf make_frag(v4u lo, v4u hi) {
    v8u r;
    r[0] = lo[0]; r[1] = lo[1]; r[2] = lo[2]; r[3] = lo[3];
    r[4] = hi[0]; r[5] = hi[1]; r[6] = hi[2]; r[7] = hi[3];
    return __builtin_bit_cast(v16bf, r);
}
__device__ inline v16bf load_afrag(const u16* p, int half) {
    v4u lo = *(const v4u*)(p + 8 * half);
    v4u hi = *(const v4u*)(p + 16 + 8 * half);
    return make_frag(lo, hi);
}
// B-fragment 32x16 bf16 (col per lane): dwords 0..3 K=8*half+{0..7}, dwords 4..7 K=16+8*half+{0..7}
__device__ inline v16bf load_bfrag(const u16* p, int half) {
    v4u lo = *(const v4u*)(p + 8 * half);
    v4u hi = *(const v4u*)(p + 16 + 8 * half);
    return make_frag(lo, hi);
}
__device__ inline v8f wmma_bf16(v16bf a, v16bf b, v8f c) {
    return __builtin_amdgcn_wmma_f32_16x16x32_bf16(false, a, false, b, (short)0, c, false, false);
}

// ---------------- CDNA5 async global->LDS copy (ASYNCcnt path) ----------------
__device__ inline void async_copy_b128(u32 lds_byte_addr, const u16* gptr) {
    asm volatile("global_load_async_to_lds_b128 %0, %1, off"
                 :: "v"(lds_byte_addr), "v"((u64)(uintptr_t)gptr)
                 : "memory");
}
__device__ inline void wait_asynccnt0() {
    asm volatile("s_wait_asynccnt 0x0" ::: "memory");
}

// ---------------- 1) x f32 -> bf16 ----------------
__global__ __launch_bounds__(256) void cvt_x_kernel(const float* __restrict__ x,
                                                    u16* __restrict__ xb) {
    int i = blockIdx.x * 256 + threadIdx.x;
    xb[i] = f2bf(x[i]);
}

// ---------------- 2) W f32 (KxN) -> bf16 transposed (NxK), LDS tiled ----------------
__global__ __launch_bounds__(256) void cvt_wT_kernel(const float* __restrict__ wq,
                                                     const float* __restrict__ wk,
                                                     const float* __restrict__ wv,
                                                     const float* __restrict__ wo,
                                                     u16* __restrict__ oq, u16* __restrict__ ok,
                                                     u16* __restrict__ ov, u16* __restrict__ oo) {
    __shared__ float tile[32][33];
    const float* W; u16* O;
    switch (blockIdx.z) {
        case 0: W = wq; O = oq; break;
        case 1: W = wk; O = ok; break;
        case 2: W = wv; O = ov; break;
        default: W = wo; O = oo; break;
    }
    int n0 = blockIdx.x * 32, k0 = blockIdx.y * 32;
    int lx = threadIdx.x & 31, ly = threadIdx.x >> 5;
    #pragma unroll
    for (int r = ly; r < 32; r += 8)
        tile[r][lx] = W[(size_t)(k0 + r) * DD + n0 + lx];
    __syncthreads();
    #pragma unroll
    for (int r = ly; r < 32; r += 8)
        O[(size_t)(n0 + r) * DD + k0 + lx] = f2bf(tile[lx][r]);
}

// ---------------- 3/6) bf16 GEMM: C[M,N] = A[M,K] * Bt[N,K]^T ----------------
// mode 0: bf16 row-major out;  mode 1: bf16 out as V^T [B][H][HD][S];  mode 2: f32 row-major out
__global__ __launch_bounds__(256) void gemm_kernel(const u16* __restrict__ A,
                                                   const u16* __restrict__ Bt,
                                                   u16* __restrict__ outB,
                                                   float* __restrict__ outF,
                                                   int mode) {
    const int lane = threadIdx.x & 31;
    const int wave = threadIdx.x >> 5;
    const int half = lane >> 4;
    const int l15  = lane & 15;
    const int m0 = blockIdx.y * 64  + (wave & 1) * 32;   // 32 rows per wave
    const int n0 = blockIdx.x * 256 + (wave >> 1) * 64;  // 64 cols per wave

    v8f acc[2][4];
    #pragma unroll
    for (int i = 0; i < 2; ++i)
        #pragma unroll
        for (int j = 0; j < 4; ++j)
            #pragma unroll
            for (int e = 0; e < 8; ++e) acc[i][j][e] = 0.0f;

    const u16* Arow  = A  + (size_t)(m0 + l15) * DD;
    const u16* Brow0 = Bt + (size_t)(n0 + l15) * DD;

    for (int k0 = 0; k0 < DD; k0 += 32) {
        v16bf a0 = load_afrag(Arow + k0, half);
        v16bf a1 = load_afrag(Arow + (size_t)16 * DD + k0, half);
        #pragma unroll
        for (int j = 0; j < 4; ++j) {
            v16bf b = load_bfrag(Brow0 + (size_t)(16 * j) * DD + k0, half);
            acc[0][j] = wmma_bf16(a0, b, acc[0][j]);
            acc[1][j] = wmma_bf16(a1, b, acc[1][j]);
        }
    }

    #pragma unroll
    for (int i = 0; i < 2; ++i) {
        #pragma unroll
        for (int j = 0; j < 4; ++j) {
            #pragma unroll
            for (int e = 0; e < 8; ++e) {
                int m = m0 + 16 * i + 8 * half + e;
                int n = n0 + 16 * j + l15;
                float v = acc[i][j][e];
                if (mode == 0) {
                    outB[(size_t)m * DD + n] = f2bf(v);
                } else if (mode == 1) {
                    int h = n >> 7, d = n & 127;           // head, dim
                    int bb = m >> 11, s = m & 2047;        // batch, seq
                    outB[(((size_t)(bb * HH + h) * HDD + d) << 11) + s] = f2bf(v);
                } else {
                    outF[(size_t)m * DD + n] = v;
                }
            }
        }
    }
}

// ---------------- 4) RoPE (in place on bf16 Q and K) ----------------
__global__ __launch_bounds__(256) void rope_kernel(u16* __restrict__ q, u16* __restrict__ k,
                                                   const float* __restrict__ fcos,
                                                   const float* __restrict__ fsin) {
    u32 t = blockIdx.x * 256 + threadIdx.x;  // pair index, 23 bits total
    u16* buf = blockIdx.y ? k : q;
    int i = t & 63;
    int h = (t >> 6) & 31;
    int s = (t >> 11) & 2047;
    int b = t >> 22;
    size_t off = (size_t)(b * SS + s) * DD + h * HDD + 2 * i;
    u32 u = *(u32*)(buf + off);
    float a = bf2f((u16)(u & 0xFFFF));
    float c = bf2f((u16)(u >> 16));
    float cv = fcos[s * HALFD + i];
    float sv = fsin[s * HALFD + i];
    *(u32*)(buf + off) = pack2bf(a * cv - c * sv, a * sv + c * cv);
}

// ---------------- 5) Flash attention (causal), K/V staged via async->LDS ----------------
// Q,K: bf16 [B][S][H*HD] (post-RoPE). Vt: bf16 [B][H][HD][S]. O: bf16 [B][S][H*HD]
__global__ __launch_bounds__(256) void attn_kernel(const u16* __restrict__ Q,
                                                   const u16* __restrict__ Kb,
                                                   const u16* __restrict__ Vt,
                                                   u16* __restrict__ O) {
    __shared__ __align__(16) u16 ldsK[32 * KLD];    // [key][d]  (+pad)
    __shared__ __align__(16) u16 ldsV[128 * VLD];   // [d][key]  (+pad)

    const int tid  = threadIdx.x;
    const int lane = tid & 31;
    const int wave = tid >> 5;
    const int half = lane >> 4;
    const int l15  = lane & 15;
    const int bh = blockIdx.y;
    const int b  = bh >> 5;
    const int h  = bh & 31;
    const int qr0 = blockIdx.x * 128 + wave * 16;  // 16 query rows per wave

    // Q^T as B operand: per lane column = its query row, contiguous along d
    const u16* qrow = Q + (size_t)(b * SS + qr0 + l15) * DD + h * HDD;
    v16bf qf[4];
    #pragma unroll
    for (int j = 0; j < 4; ++j) qf[j] = load_bfrag(qrow + 32 * j, half);

    v8f of[8];
    #pragma unroll
    for (int dt = 0; dt < 8; ++dt)
        #pragma unroll
        for (int e = 0; e < 8; ++e) of[dt][e] = 0.0f;

    float m_r = -1e30f, l_r = 0.0f;
    const u16* kbase = Kb + (size_t)(b * SS) * DD + h * HDD;
    const u16* vbase = Vt + (size_t)(b * HH + h) * HDD * SS;
    const float scale = 0.088388347648318447f;  // 1/sqrt(128)
    const int qg = qr0 + l15;                   // this lane's query row

    const int nkb = blockIdx.x * 4 + 4;         // block-uniform causal bound (32 keys per iter)
    for (int it = 0; it < nkb; ++it) {
        const int kb0 = it * 32;
        __syncthreads();  // previous iteration's LDS reads complete
        // ---- stage K tile: 32 rows x 128 d  (512 b128 transfers, 2 per thread)
        #pragma unroll
        for (int r = 0; r < 2; ++r) {
            int i = tid + 256 * r;
            int row = i >> 4, ch = i & 15;
            const u16* gp = kbase + (size_t)(kb0 + row) * DD + ch * 8;
            u32 ld = (u32)(uintptr_t)&ldsK[row * KLD + ch * 8];
            async_copy_b128(ld, gp);
        }
        // ---- stage V^T tile: 128 rows x 32 keys (512 b128 transfers, 2 per thread)
        #pragma unroll
        for (int r = 0; r < 2; ++r) {
            int j = tid + 256 * r;
            int d = j >> 2, ch = j & 3;
            const u16* gp = vbase + (size_t)d * SS + kb0 + ch * 8;
            u32 ld = (u32)(uintptr_t)&ldsV[d * VLD + ch * 8];
            async_copy_b128(ld, gp);
        }
        wait_asynccnt0();
        __syncthreads();  // all waves' async copies visible

        // ---- S^T tiles from LDS: D[key, qrow] = K_tile x Q^T ; lane owns one query row
        v8f st[2];
        #pragma unroll
        for (int t = 0; t < 2; ++t) {
            #pragma unroll
            for (int e = 0; e < 8; ++e) st[t][e] = 0.0f;
            const u16* krow = &ldsK[(16 * t + l15) * KLD];
            #pragma unroll
            for (int j = 0; j < 4; ++j) {
                v16bf kf = load_afrag(krow + 32 * j, half);
                st[t] = wmma_bf16(kf, qf[j], st[t]);
            }
        }
        // ---- per-lane online softmax over this lane's 16 scores
        float sv[2][8];
        float mloc = m_r;
        #pragma unroll
        for (int t = 0; t < 2; ++t)
            #pragma unroll
            for (int e = 0; e < 8; ++e) {
                int kg = kb0 + 16 * t + 8 * half + e;
                float x = st[t][e] * scale + (kg > qg ? -1e9f : 0.0f);
                sv[t][e] = x;
                mloc = fmaxf(mloc, x);
            }
        float alpha = __builtin_amdgcn_exp2f((m_r - mloc) * LOG2E);
        float psum = 0.0f;
        float pf[2][8];
        #pragma unroll
        for (int t = 0; t < 2; ++t)
            #pragma unroll
            for (int e = 0; e < 8; ++e) {
                float p = __builtin_amdgcn_exp2f((sv[t][e] - mloc) * LOG2E);
                pf[t][e] = p;
                psum += p;
            }
        l_r = l_r * alpha + psum;
        m_r = mloc;

        // ---- P A-fragment: element-for-element concat of the two S^T C-fragments
        v8u pk;
        #pragma unroll
        for (int i = 0; i < 4; ++i) {
            pk[i]     = pack2bf(pf[0][2 * i], pf[0][2 * i + 1]);
            pk[4 + i] = pack2bf(pf[1][2 * i], pf[1][2 * i + 1]);
        }
        v16bf pA = __builtin_bit_cast(v16bf, pk);

        // ---- broadcast per-row rescale factors (O frag rows = e + 8*half)
        float fa[8];
        #pragma unroll
        for (int e = 0; e < 8; ++e) fa[e] = __shfl(alpha, half * 8 + e, 32);

        #pragma unroll
        for (int dt = 0; dt < 8; ++dt) {
            const u16* vrow = &ldsV[(dt * 16 + l15) * VLD];
            v16bf vf = load_bfrag(vrow, half);
            #pragma unroll
            for (int e = 0; e < 8; ++e) of[dt][e] *= fa[e];
            of[dt] = wmma_bf16(pA, vf, of[dt]);
        }
    }

    float fi[8];
    #pragma unroll
    for (int e = 0; e < 8; ++e) fi[e] = __builtin_amdgcn_rcpf(__shfl(l_r, half * 8 + e, 32));
    #pragma unroll
    for (int dt = 0; dt < 8; ++dt)
        #pragma unroll
        for (int e = 0; e < 8; ++e) {
            int m = qr0 + 8 * half + e;
            O[(size_t)(b * SS + m) * DD + h * HDD + dt * 16 + l15] = f2bf(of[dt][e] * fi[e]);
        }
}

// ---------------- host launch ----------------
extern "C" void kernel_launch(void* const* d_in, const int* in_sizes, int n_in,
                              void* d_out, int out_size, void* d_ws, size_t ws_size,
                              hipStream_t stream) {
    const float* x    = (const float*)d_in[0];
    const float* Wq   = (const float*)d_in[1];
    const float* Wk   = (const float*)d_in[2];
    const float* Wv   = (const float*)d_in[3];
    const float* Wo   = (const float*)d_in[4];
    const float* fcos = (const float*)d_in[5];
    const float* fsin = (const float*)d_in[6];
    float* out = (float*)d_out;

    const size_t sz = (size_t)BB * SS * DD * sizeof(u16);  // 32 MiB (same as one DxD bf16 weight)
    char* w = (char*)d_ws;
    u16* xb  = (u16*)(w + 0 * sz);
    u16* wqT = (u16*)(w + 1 * sz);
    u16* wkT = (u16*)(w + 2 * sz);
    u16* wvT = (u16*)(w + 3 * sz);
    u16* woT = (u16*)(w + 4 * sz);
    u16* qb  = (u16*)(w + 5 * sz);
    u16* kb  = (u16*)(w + 6 * sz);
    u16* vT  = (u16*)(w + 7 * sz);
    u16* ao  = (u16*)(w + 8 * sz);

    // 1) x -> bf16
    cvt_x_kernel<<<(BB * SS * DD) / 256, 256, 0, stream>>>(x, xb);
    // 2) weights -> bf16, transposed
    cvt_wT_kernel<<<dim3(DD / 32, DD / 32, 4), 256, 0, stream>>>(Wq, Wk, Wv, Wo, wqT, wkT, wvT, woT);
    // 3) QKV projections (V stored transposed per head)
    dim3 ggrid(DD / 256, (BB * SS) / 64);
    gemm_kernel<<<ggrid, 256, 0, stream>>>(xb, wqT, qb, nullptr, 0);
    gemm_kernel<<<ggrid, 256, 0, stream>>>(xb, wkT, kb, nullptr, 0);
    gemm_kernel<<<ggrid, 256, 0, stream>>>(xb, wvT, vT, nullptr, 1);
    // 4) RoPE on Q and K
    rope_kernel<<<dim3((BB * SS * HH * HALFD) / 256, 2), 256, 0, stream>>>(qb, kb, fcos, fsin);
    // 5) causal flash attention (async LDS staging)
    attn_kernel<<<dim3(SS / 128, BB * HH), 256, 0, stream>>>(qb, kb, vT, ao);
    // 6) output projection -> f32
    gemm_kernel<<<ggrid, 256, 0, stream>>>(ao, woT, nullptr, out, 2);
}
